// mHC_37933151158951
// MI455X (gfx1250) — matrix operand order repaired and checked by reading, not compiled, not run on hardware
//
#include <hip/hip_runtime.h>

#define NSY 4
#define DD 1024
#define KK 4096
#define TOK_PER_BLK 16
#define EPS_RMS 1.1920928955078125e-07f
#define SINK_EPS 1e-12f
#define NFRAG_USHORT (2 * 128 * 32 * 16)  // 131072 bf16 = 256 KB swizzled weights

typedef __attribute__((ext_vector_type(16))) __bf16 v16bf;
typedef __attribute__((ext_vector_type(8))) float v8f;

union BFPack { v16bf v; __bf16 h[16]; unsigned short u[16]; };

__device__ __forceinline__ unsigned short f2bf_bits(float f) {
  unsigned int u = __float_as_uint(f);
  u += 0x7FFFu + ((u >> 16) & 1u);  // round-to-nearest-even
  return (unsigned short)(u >> 16);
}

// ---------------------------------------------------------------------------
// Prepass: bf16(W_combined * norm_w) written in this kernel's WMMA-B lane order (contiguous k):
// wbf[((t*128 + kb)*32 + lane)*16 + j]  where lane-half picks the 16-wide K
// block of fragment kb (K = kb*32 .. kb*32+31), lane&15 = output column.
// ---------------------------------------------------------------------------
__global__ __launch_bounds__(256) void pack_weights(
    const float* __restrict__ Wpre, const float* __restrict__ Wpost,
    const float* __restrict__ Wres, const float* __restrict__ nw,
    unsigned short* __restrict__ wbf) {
  const int idx = blockIdx.x * 256 + threadIdx.x;  // < 131072
  const int j = idx & 15;
  const int lane = (idx >> 4) & 31;
  const int kb = (idx >> 9) & 127;
  const int t = idx >> 16;
  const int half = lane >> 4, l15 = lane & 15;
  const int nglob = t * 16 + l15;
  const int kg = kb * 32 + 16 * half + j;
  float v = 0.f;
  if (nglob < 4)       v = Wpre[nglob * KK + kg];
  else if (nglob < 8)  v = Wpost[(nglob - 4) * KK + kg];
  else if (nglob < 24) v = Wres[(nglob - 8) * KK + kg];
  v *= nw[kg & (DD - 1)];
  wbf[idx] = f2bf_bits(v);
}

// ---------------------------------------------------------------------------
// Fused kernel: 16 tokens per 256-thread block (8 wave32).
// ---------------------------------------------------------------------------
template <int PREP>
__global__ __launch_bounds__(256) void mhc_fused(
    const float* __restrict__ x, const float* __restrict__ nw,
    const float* __restrict__ Wpre, const float* __restrict__ Wpost,
    const float* __restrict__ Wres, const float* __restrict__ apre,
    const float* __restrict__ apost, const float* __restrict__ ares,
    const unsigned short* __restrict__ wbf, float* __restrict__ out) {
  __shared__ float ssq[64];             // [token16][stream4]
  __shared__ float invr[64];
  __shared__ float hpart[8 * 16 * 32];  // [wave][tile*8+i][lane]  (16 KB)
  __shared__ float hmat[16 * 24];
  __shared__ float wpreS[16 * 4];
  __shared__ float wpostS[16 * 4];
  __shared__ float wresS[16 * 16];

  const int tid = threadIdx.x;
  const int w = tid >> 5, lane = tid & 31;
  const long long base = (long long)blockIdx.x * (TOK_PER_BLK * KK);

  // ------------- Phase 0: sum-of-squares, 4 threads per (token,stream)
  {
    const float4* x4 = (const float4*)(x + base);
    const int p = tid >> 2, sub = tid & 3;   // pair = token*4+stream
    const int b0 = p * 256 + sub;
    float s = 0.f;
    for (int it = 0; it < 64; ++it) {
      float4 v = x4[b0 + it * 4];
      s = fmaf(v.x, v.x, s); s = fmaf(v.y, v.y, s);
      s = fmaf(v.z, v.z, s); s = fmaf(v.w, v.w, s);
    }
    s += __shfl_xor(s, 1, 32);
    s += __shfl_xor(s, 2, 32);
    if (sub == 0) ssq[p] = s;
  }
  __syncthreads();
  if (tid < 64) invr[tid] = rsqrtf(ssq[tid] * (1.0f / DD) + EPS_RMS);
  __syncthreads();

  // ------------- Phase 1: projections. Wave w owns K in [w*512, w*512+512),
  // builds A once per K-block and feeds BOTH 16-wide N-tiles.
  {
    const int q = w >> 1;  // stream index of this wave's K-slice
    const int half = lane >> 4, l15 = lane & 15;
    const float* xrow = x + base + (long long)l15 * KK;
    const float* Wrow0 = Wres; const float* Wrow1 = Wres; float wmask1 = 1.f;
    if (!PREP) {
      const int n0 = l15;
      Wrow0 = (n0 < 4) ? Wpre + n0 * KK
            : (n0 < 8) ? Wpost + (n0 - 4) * KK : Wres + (n0 - 8) * KK;
      const int n1 = 16 + l15;
      if (n1 < 24) Wrow1 = Wres + (n1 - 8) * KK;
      else         wmask1 = 0.f;
    }
    v8f acc0 = {}, acc1 = {};
    for (int kki = 0; kki < 16; ++kki) {
      const int kb = w * 16 + kki;  // global 32-wide K block
      BFPack A, B0, B1;
#pragma unroll
      for (int v2 = 0; v2 < 8; ++v2) {
        int k0 = kb * 32 + 16 * half + 2 * v2;
        float2 f = *(const float2*)(xrow + k0);
        A.h[2 * v2]     = (__bf16)f.x;
        A.h[2 * v2 + 1] = (__bf16)f.y;
      }
      if (PREP) {
        B0.v = *(const v16bf*)(wbf + ((0 * 128 + kb) * 32 + lane) * 16);
        B1.v = *(const v16bf*)(wbf + ((1 * 128 + kb) * 32 + lane) * 16);
      } else {
        const int kg = kb * 32 + 16 * half;
        const int kn = kg & (DD - 1);
#pragma unroll
        for (int j = 0; j < 16; ++j) {
          float nwj = nw[kn + j];
          B0.h[j] = (__bf16)(Wrow0[kg + j] * nwj);
          B1.h[j] = (__bf16)(Wrow1[kg + j] * nwj * wmask1);
        }
      }
      acc0 = __builtin_amdgcn_wmma_f32_16x16x32_bf16(false, A.v, false, B0.v,
                                                     (short)0, acc0, false, false);
      acc1 = __builtin_amdgcn_wmma_f32_16x16x32_bf16(false, A.v, false, B1.v,
                                                     (short)0, acc1, false, false);
    }
#pragma unroll
    for (int i = 0; i < 8; ++i) {
      const int m = i + 8 * half;
      const float sc = invr[m * 4 + q];
      hpart[(w * 16 + i) * 32 + lane]       = acc0[i] * sc;
      hpart[(w * 16 + 8 + i) * 32 + lane]   = acc1[i] * sc;
    }
  }
  // Prefetch this thread's phase-3 lines while we sit behind barriers
  {
    const int tk = tid >> 4, g = tid & 15;
    const float* a = x + base + (long long)tk * KK + g * 4;
#pragma unroll
    for (int j = 0; j < NSY; ++j) __builtin_prefetch(a + j * DD, 0, 0);
  }
  __syncthreads();

  // ------------- Phase 1b: reduce the 8 K-slices, undo C lane layout
  for (int r = 0; r < 2; ++r) {
    const int e = tid + r * 256;          // [tile][i][lane]
    const int tt = e >> 8, i = (e >> 5) & 7, ln = e & 31;
    float s = 0.f;
#pragma unroll
    for (int ww = 0; ww < 8; ++ww) s += hpart[(ww * 16 + tt * 8 + i) * 32 + ln];
    const int m = i + 8 * (ln >> 4), n = (ln & 15) + 16 * tt;
    if (n < 24) hmat[m * 24 + n] = s;
  }
  __syncthreads();

  // ------------- Phase 2: activations + 4x4 Sinkhorn (one thread per token)
  if (tid < 16) {
    const float ap = apre[0], apo = apost[0], ar = ares[0];
    const float* hm = hmat + tid * 24;
    float Am[16];
#pragma unroll
    for (int j = 0; j < 4; ++j)
      wpreS[tid * 4 + j] = 1.0f / (1.0f + expf(-(ap + tanhf(hm[j]))));
#pragma unroll
    for (int i = 0; i < 4; ++i)
      wpostS[tid * 4 + i] = 2.0f / (1.0f + expf(-(apo + tanhf(hm[4 + i]))));
#pragma unroll
    for (int e2 = 0; e2 < 16; ++e2)
      Am[e2] = expf(ar + tanhf(hm[8 + e2]));
    for (int itn = 0; itn < 20; ++itn) {
      for (int i = 0; i < 4; ++i) {
        float rs = Am[i*4] + Am[i*4+1] + Am[i*4+2] + Am[i*4+3] + SINK_EPS;
        float inv = 1.0f / rs;
        for (int j = 0; j < 4; ++j) Am[i*4+j] *= inv;
      }
      for (int j = 0; j < 4; ++j) {
        float cs = Am[j] + Am[4+j] + Am[8+j] + Am[12+j] + SINK_EPS;
        float inv = 1.0f / cs;
        for (int i = 0; i < 4; ++i) Am[i*4+j] *= inv;
      }
    }
#pragma unroll
    for (int e2 = 0; e2 < 16; ++e2) wresS[tid * 16 + e2] = Am[e2];
  }
  __syncthreads();

  // ------------- Phase 3: out = w_res . x + w_post (x) (w_pre . x)
  {
    const int tk = tid >> 4, g = tid & 15;
    float wp[4], wq[4], wr[16];
#pragma unroll
    for (int j = 0; j < 4; ++j) { wp[j] = wpreS[tk*4+j]; wq[j] = wpostS[tk*4+j]; }
#pragma unroll
    for (int e2 = 0; e2 < 16; ++e2) wr[e2] = wresS[tk*16+e2];
    const float4* xt = (const float4*)(x + base + (long long)tk * KK);
    float4* ot = (float4*)(out + base + (long long)tk * KK);
    for (int it2 = 0; it2 < 16; ++it2) {
      const int d4 = it2 * 16 + g;
      float4 xv[4], ov[4];
#pragma unroll
      for (int j = 0; j < 4; ++j) xv[j] = xt[j * 256 + d4];  // L2 hits
      const float* xf = (const float*)xv;
      float* of = (float*)ov;
#pragma unroll
      for (int c = 0; c < 4; ++c) {
        float y = wp[0]*xf[c] + wp[1]*xf[4+c] + wp[2]*xf[8+c] + wp[3]*xf[12+c];
#pragma unroll
        for (int i = 0; i < 4; ++i)
          of[i*4+c] = wr[i*4+0]*xf[c] + wr[i*4+1]*xf[4+c]
                    + wr[i*4+2]*xf[8+c] + wr[i*4+3]*xf[12+c] + wq[i]*y;
      }
#pragma unroll
      for (int i = 0; i < 4; ++i) ot[i * 256 + d4] = ov[i];  // coalesced
    }
  }
}

extern "C" void kernel_launch(void* const* d_in, const int* in_sizes, int n_in,
                              void* d_out, int out_size, void* d_ws, size_t ws_size,
                              hipStream_t stream) {
  const float* x     = (const float*)d_in[0];
  const float* nw    = (const float*)d_in[1];
  const float* Wpre  = (const float*)d_in[2];
  const float* Wpost = (const float*)d_in[3];
  const float* Wres  = (const float*)d_in[4];
  const float* apre  = (const float*)d_in[5];
  const float* apost = (const float*)d_in[6];
  const float* ares  = (const float*)d_in[7];
  float* out = (float*)d_out;
  const int tokens = in_sizes[0] / (NSY * DD);
  const int nblk = tokens / TOK_PER_BLK;
  if (ws_size >= NFRAG_USHORT * sizeof(unsigned short)) {
    unsigned short* wbf = (unsigned short*)d_ws;
    pack_weights<<<NFRAG_USHORT / 256, 256, 0, stream>>>(Wpre, Wpost, Wres, nw, wbf);
    mhc_fused<1><<<nblk, 256, 0, stream>>>(x, nw, Wpre, Wpost, Wres,
                                           apre, apost, ares, wbf, out);
  } else {
    mhc_fused<0><<<nblk, 256, 0, stream>>>(x, nw, Wpre, Wpost, Wres,
                                           apre, apost, ares, nullptr, out);
  }
}
